// GAU_60473139527723
// MI455X (gfx1250) — compile-verified, matrix operand order checked by reading, not run on hardware
//
#include <hip/hip_runtime.h>
#include <hip/hip_bf16.h>
#include <cstdint>

// ---------------------------------------------------------------------------
// GAU forward for MI455X (gfx1250).
// All GEMMs: v_wmma_f32_16x16x32_bf16 with double-buffered LDS staging fed by
// GLOBAL_LOAD_ASYNC_TO_LDS_B128 (ASYNCcnt-tracked async memory->LDS copies).
// Pipeline:
//   1) LayerNorm(x) -> normed (bf16)
//   2) convert W_hidden / W_qk / W_out to bf16
//   3) GEMM  hg = silu(normed @ Wh^T + bh)   -> vT (bf16, transposed), gate
//   4) GEMM  Z  = silu(normed @ Wqk^T + bqk) -> q, k (bf16, affine)
//   5) GEMM  A  = relu(q @ k^T / S)^2        -> A (bf16, per batch)
//   6) GEMM  Vg = (A @ vT^T) * gate          -> Vg (bf16)
//   7) GEMM  out = Vg @ Wout^T + bout + x    -> f32 output
// Every B operand is [N x K] row-major, so one uniform async pipeline.
// ---------------------------------------------------------------------------

typedef __attribute__((ext_vector_type(16))) __bf16 bf16x16;
typedef __attribute__((ext_vector_type(8)))  float  f32x8;

#define BM 128
#define BN 128
#define BK 32
#define ASTR (BK + 8)   // LDS row stride (bf16 elems) = 80B, keeps 16B alignment
#define BSTR (BK + 8)

union FragU { bf16x16 v; uint4 q[2]; };

// Async memory->LDS copy, 16B per lane. lds = 32-bit LDS byte address,
// gptr = per-lane global address. Tracked by ASYNCcnt.
__device__ __forceinline__ void async_copy_b128(uint32_t lds, const void* gptr) {
  asm volatile("global_load_async_to_lds_b128 %0, %1, off"
               :: "v"(lds), "v"(gptr) : "memory");
}

// Build a 16x(K=32) WMMA fragment from an LDS tile stored [row][k] row-major.
// Per ISA 7.12.2 (16-bit A/B): lane<16 holds row=lane, k 0..7 / 16..23;
// lane>=16 holds row=lane-16, k 8..15 / 24..31.
__device__ __forceinline__ bf16x16 load_frag(const __bf16* base, int stride) {
  int lane = threadIdx.x & 31;
  int r = lane & 15;
  int h = (lane >> 4) * 8;
  const __bf16* p = base + r * stride + h;
  FragU f;
  f.q[0] = *(const uint4*)(p);
  f.q[1] = *(const uint4*)(p + 16);
  return f.v;
}

__device__ __forceinline__ float silu_f(float y) {
  return y / (1.0f + __expf(-y));
}

// ---------------------------------------------------------------------------
// Tiled bf16 GEMM, C[M x N] = A[M x K](row) * B[N x K](row)^T, fused epilogue.
// EPI: 0 = silu + bias; col < splitN -> vT[b][n][m'] (transposed), else gate
//      1 = silu + bias, two affine outputs q = y*g0+b0, k = y*g1+b1
//      2 = relu(acc*scale)^2 -> bf16
//      3 = acc * gate[m,n]   -> bf16
//      4 = acc + bias[n] + resid[m,n] -> f32
// ---------------------------------------------------------------------------
template <int EPI>
__global__ __launch_bounds__(256)
void gau_gemm(const __bf16* __restrict__ A, long long sA, int lda,
              const __bf16* __restrict__ B, long long sB, int ldb,
              int K,
              const float* __restrict__ bias,
              const float* __restrict__ g0, const float* __restrict__ b0,
              const float* __restrict__ g1, const float* __restrict__ b1,
              const __bf16* __restrict__ gate, long long sG, int ldg,
              const float* __restrict__ resid, int ldr,
              void* __restrict__ out0, long long sO0, int ldo0,
              void* __restrict__ out1, int ldo1,
              float scale, int splitN) {
  __shared__ __attribute__((aligned(16))) __bf16 AsmS[2][BM * ASTR];
  __shared__ __attribute__((aligned(16))) __bf16 BsmS[2][BN * BSTR];

  const int tid = threadIdx.x;
  const size_t zb = blockIdx.z;
  const __bf16* Ab = A + zb * (size_t)sA;
  const __bf16* Bb = B + zb * (size_t)sB;
  const int m_blk = blockIdx.y * BM;
  const int n_blk = blockIdx.x * BN;

  const int wid = tid >> 5;
  const int wm  = wid & 3;       // wave M tile: 4 waves * 32 rows
  const int wn  = wid >> 2;      // wave N tile: 2 waves * 64 cols

  const f32x8 zero = {0.f, 0.f, 0.f, 0.f, 0.f, 0.f, 0.f, 0.f};
  f32x8 acc[2][4];
#pragma unroll
  for (int i = 0; i < 2; ++i)
#pragma unroll
    for (int j = 0; j < 4; ++j) acc[i][j] = zero;

  // cooperative-load indices: 256 threads, 2 x 16B per tile each
  const int a_r = tid >> 1;          // 0..127
  const int a_c = (tid & 1) * 16;    // 0 or 16

  // raw 32-bit LDS byte addresses for the async copies
  const uint32_t ldsA = (uint32_t)(uintptr_t)&AsmS[0][0];
  const uint32_t ldsB = (uint32_t)(uintptr_t)&BsmS[0][0];
  const uint32_t aoff = (uint32_t)((a_r * ASTR + a_c) * 2);
  const uint32_t boff = (uint32_t)((a_r * BSTR + a_c) * 2);
  const uint32_t bufA = (uint32_t)(BM * ASTR * 2);
  const uint32_t bufB = (uint32_t)(BN * BSTR * 2);

  auto prefetch = [&](int buf, int k0) {
    const __bf16* ap = Ab + (size_t)(m_blk + a_r) * lda + k0 + a_c;
    const __bf16* bp = Bb + (size_t)(n_blk + a_r) * ldb + k0 + a_c;
    const uint32_t la = ldsA + (uint32_t)buf * bufA + aoff;
    const uint32_t lb = ldsB + (uint32_t)buf * bufB + boff;
    async_copy_b128(la,      ap);
    async_copy_b128(la + 16, ap + 8);
    async_copy_b128(lb,      bp);
    async_copy_b128(lb + 16, bp + 8);
  };

  const int nstages = K / BK;
  prefetch(0, 0);
  for (int st = 0; st < nstages; ++st) {
    const int buf = st & 1;
    if (st + 1 < nstages) {
      prefetch(buf ^ 1, (st + 1) * BK);
      // 4 ops just issued for the next stage may stay in flight; the previous
      // 4 (current buffer) are guaranteed complete (in-order async loads).
      asm volatile("s_wait_asynccnt 0x4" ::: "memory");
    } else {
      asm volatile("s_wait_asynccnt 0x0" ::: "memory");
    }
    __syncthreads();   // all waves' async copies for `buf` have landed

    const __bf16* Abase = &AsmS[buf][0];
    const __bf16* Bbase = &BsmS[buf][0];
    bf16x16 af0 = load_frag(Abase + (wm * 32 + 0)  * ASTR, ASTR);
    bf16x16 af1 = load_frag(Abase + (wm * 32 + 16) * ASTR, ASTR);
#pragma unroll
    for (int ni = 0; ni < 4; ++ni) {
      bf16x16 bfr = load_frag(Bbase + (wn * 64 + ni * 16) * BSTR, BSTR);
      acc[0][ni] = __builtin_amdgcn_wmma_f32_16x16x32_bf16(
          false, af0, false, bfr, (short)0, acc[0][ni], false, false);
      acc[1][ni] = __builtin_amdgcn_wmma_f32_16x16x32_bf16(
          false, af1, false, bfr, (short)0, acc[1][ni], false, false);
    }
    __syncthreads();   // done reading `buf`; next stage may overwrite it
  }

  // ---- epilogue: C layout (ISA 7.12.2): lane<16 -> N=lane, M=vgpr r;
  //      lane>=16 -> N=lane-16, M=r+8 ----
  const int lane = tid & 31;
  const int cn = lane & 15;
  const int mh = (lane >> 4) * 8;
#pragma unroll
  for (int mi = 0; mi < 2; ++mi) {
    const int mrow0 = m_blk + wm * 32 + mi * 16 + mh;
#pragma unroll
    for (int ni = 0; ni < 4; ++ni) {
      const int n = n_blk + wn * 64 + ni * 16 + cn;
      f32x8 a = acc[mi][ni];
#pragma unroll
      for (int e = 0; e < 8; ++e) {
        const int m = mrow0 + e;
        float y = a[e];
        if constexpr (EPI == 0) {
          y = silu_f(y + bias[n]);
          if (n < splitN) {
            // store v transposed: vT[batch][n][m'] with ldo0 == S
            const int b_  = m / ldo0;
            const int mp  = m - b_ * ldo0;
            ((__bf16*)out0)[((size_t)b_ * splitN + n) * ldo0 + mp] = (__bf16)y;
          } else {
            ((__bf16*)out1)[(size_t)m * ldo1 + (n - splitN)] = (__bf16)y;
          }
        } else if constexpr (EPI == 1) {
          y = silu_f(y + bias[n]);
          ((__bf16*)out0)[(size_t)m * ldo0 + n] = (__bf16)(y * g0[n] + b0[n]);
          ((__bf16*)out1)[(size_t)m * ldo1 + n] = (__bf16)(y * g1[n] + b1[n]);
        } else if constexpr (EPI == 2) {
          y *= scale;
          y = (y > 0.f) ? y * y : 0.f;
          ((__bf16*)out0)[zb * (size_t)sO0 + (size_t)m * ldo0 + n] = (__bf16)y;
        } else if constexpr (EPI == 3) {
          float g = (float)gate[zb * (size_t)sG + (size_t)m * ldg + n];
          ((__bf16*)out0)[zb * (size_t)sO0 + (size_t)m * ldo0 + n] = (__bf16)(y * g);
        } else {  // EPI == 4
          y += bias[n] + resid[(size_t)m * ldr + n];
          ((float*)out0)[(size_t)m * ldo0 + n] = y;
        }
      }
    }
  }
}

// ---------------------------------------------------------------------------
// LayerNorm (D=768), one block per row, output bf16.
// ---------------------------------------------------------------------------
__global__ __launch_bounds__(256)
void gau_ln(const float* __restrict__ x, const float* __restrict__ w,
            const float* __restrict__ b, __bf16* __restrict__ out) {
  const int D = 768;
  const int row = blockIdx.x;
  const float* xr = x + (size_t)row * D;
  float v[3], s = 0.f, s2 = 0.f;
#pragma unroll
  for (int i = 0; i < 3; ++i) {
    v[i] = xr[threadIdx.x + i * 256];
    s += v[i];
    s2 += v[i] * v[i];
  }
  __shared__ float r1[256], r2[256];
  r1[threadIdx.x] = s;
  r2[threadIdx.x] = s2;
  __syncthreads();
  for (int off = 128; off > 0; off >>= 1) {
    if (threadIdx.x < off) {
      r1[threadIdx.x] += r1[threadIdx.x + off];
      r2[threadIdx.x] += r2[threadIdx.x + off];
    }
    __syncthreads();
  }
  const float mu = r1[0] * (1.0f / D);
  const float var = r2[0] * (1.0f / D) - mu * mu;
  const float rs = rsqrtf(var + 1e-5f);
#pragma unroll
  for (int i = 0; i < 3; ++i) {
    const int c = threadIdx.x + i * 256;
    out[(size_t)row * D + c] = (__bf16)((v[i] - mu) * rs * w[c] + b[c]);
  }
}

// fp32 -> bf16 weight conversion
__global__ __launch_bounds__(256)
void gau_cvt(const float* __restrict__ in, __bf16* __restrict__ out, int n) {
  const int i = blockIdx.x * 256 + threadIdx.x;
  if (i < n) out[i] = (__bf16)in[i];
}

// ---------------------------------------------------------------------------
extern "C" void kernel_launch(void* const* d_in, const int* in_sizes, int n_in,
                              void* d_out, int out_size, void* d_ws, size_t ws_size,
                              hipStream_t stream) {
  const float* x     = (const float*)d_in[0];
  const float* ln_w  = (const float*)d_in[1];
  const float* ln_b  = (const float*)d_in[2];
  const float* Wh    = (const float*)d_in[3];
  const float* bh    = (const float*)d_in[4];
  const float* Wqk   = (const float*)d_in[5];
  const float* bqk   = (const float*)d_in[6];
  const float* gamma = (const float*)d_in[7];
  const float* beta  = (const float*)d_in[8];
  const float* Wout  = (const float*)d_in[9];
  const float* bout  = (const float*)d_in[10];

  constexpr int Bsz = 4, S = 2048, D = 768, H = 1536, QK = 128;
  const int M = Bsz * S;  // 8192

  char* base = (char*)d_ws;
  size_t off = 0;
  auto alloc = [&](size_t nelem) -> __bf16* {
    __bf16* p = (__bf16*)(base + off);
    off += (nelem * 2 + 255) & ~(size_t)255;
    return p;
  };
  __bf16* nrm  = alloc((size_t)M * D);        // normed
  __bf16* WhB  = alloc((size_t)2 * H * D);    // W_hidden bf16
  __bf16* WqkB = alloc((size_t)QK * D);       // W_qk bf16
  __bf16* WoB  = alloc((size_t)D * H);        // W_out bf16
  __bf16* vT   = alloc((size_t)Bsz * H * S);  // v, transposed per batch [H][S]
  __bf16* gbuf = alloc((size_t)M * H);        // gate [M][H]
  __bf16* qbuf = alloc((size_t)M * QK);       // q
  __bf16* kbuf = alloc((size_t)M * QK);       // k
  __bf16* Abuf = alloc((size_t)Bsz * S * S);  // attention matrix
  __bf16* Vg   = alloc((size_t)M * H);        // (A@v)*gate
  (void)ws_size; (void)in_sizes; (void)n_in; (void)out_size;

  dim3 blk(256);

  // 1) LayerNorm
  gau_ln<<<M, blk, 0, stream>>>(x, ln_w, ln_b, nrm);

  // 2) weight conversions (all counts divisible by 256)
  gau_cvt<<<(2 * H * D) / 256, blk, 0, stream>>>(Wh, WhB, 2 * H * D);
  gau_cvt<<<(QK * D) / 256, blk, 0, stream>>>(Wqk, WqkB, QK * D);
  gau_cvt<<<(D * H) / 256, blk, 0, stream>>>(Wout, WoB, D * H);

  // 3) hidden projection + SiLU -> vT (transposed) / gate
  gau_gemm<0><<<dim3((2 * H) / BN, M / BM, 1), blk, 0, stream>>>(
      nrm, 0, D, WhB, 0, D, D,
      bh, nullptr, nullptr, nullptr, nullptr,
      nullptr, 0, 0, nullptr, 0,
      vT, 0, S, gbuf, H, 1.0f, H);

  // 4) qk projection + SiLU + per-head affine -> q, k
  gau_gemm<1><<<dim3(QK / BN, M / BM, 1), blk, 0, stream>>>(
      nrm, 0, D, WqkB, 0, D, D,
      bqk, gamma, beta, gamma + QK, beta + QK,
      nullptr, 0, 0, nullptr, 0,
      qbuf, 0, QK, kbuf, QK, 1.0f, 0);

  // 5) sim = q @ k^T / S, A = relu(sim)^2   (batched over B)
  gau_gemm<2><<<dim3(S / BN, S / BM, Bsz), blk, 0, stream>>>(
      qbuf, (long long)S * QK, QK, kbuf, (long long)S * QK, QK, QK,
      nullptr, nullptr, nullptr, nullptr, nullptr,
      nullptr, 0, 0, nullptr, 0,
      Abuf, (long long)S * S, S, nullptr, 0, 1.0f / (float)S, 0);

  // 6) Vg = (A @ v) * gate   (batched; vT is [H x S] row-major = N x K)
  gau_gemm<3><<<dim3(H / BN, S / BM, Bsz), blk, 0, stream>>>(
      Abuf, (long long)S * S, S, vT, (long long)H * S, S, S,
      nullptr, nullptr, nullptr, nullptr, nullptr,
      gbuf, (long long)S * H, H, nullptr, 0,
      Vg, (long long)S * H, H, nullptr, 0, 1.0f, 0);

  // 7) out = Vg @ Wout^T + bout + x  (batch folded into M)
  gau_gemm<4><<<dim3(D / BN, M / BM, 1), blk, 0, stream>>>(
      Vg, 0, H, WoB, 0, H, H,
      bout, nullptr, nullptr, nullptr, nullptr,
      nullptr, 0, 0, x, D,
      d_out, 0, D, nullptr, 0, 1.0f, 0);
}
